// PerformerEncoderOnlyModule_18081812316567
// MI455X (gfx1250) — matrix operand order repaired and checked by reading, not compiled, not run on hardware
//
#include <hip/hip_runtime.h>
#include <cstdint>

// ---------------- model dims ----------------
#define E_      256
#define DH_     64
#define INNER_  512
#define L_      8
#define S_      2048
#define N_      4
#define TOK_    (N_ * S_)      // 8192
#define VOC_    257
#define NBF_    266
#define FPAD_   288            // 266 padded to a multiple of 32 (pad features are exact zeros)
#define HGLB_   4
#define HLOC_   4
#define NWIN_   8
#define FF_     1024
#define FF2_    2048
#define CH_     128
#define NCH_    16

typedef __bf16 bf16_t;
typedef __attribute__((ext_vector_type(16))) __bf16 v16bf;
typedef __attribute__((ext_vector_type(8)))  float  v8f;

struct alignas(16) U128 { uint32_t a, b, c, d; };
union FragU { U128 q[2]; v16bf v; };

__device__ __forceinline__ v8f wmma_bf16(v16bf a, v16bf b, v8f c) {
  return __builtin_amdgcn_wmma_f32_16x16x32_bf16(false, a, false, b, (short)0, c,
                                                 false, false);
}

// A fragment 16x32, row-major source: elem(m,k) at base[m*stride + k]
__device__ __forceinline__ v16bf frag_a_row(const bf16_t* base, int stride) {
  const int lane = threadIdx.x & 31;
  const bf16_t* p = base + (lane & 15) * stride + ((lane >> 4) << 3);
  FragU u;
  u.q[0] = *(const U128*)(p);
  u.q[1] = *(const U128*)(p + 16);
  return u.v;
}

// B fragment 32x16 from N-major storage: elem B(k,n) at base[n*stride + k]
__device__ __forceinline__ v16bf frag_b_nmaj(const bf16_t* base, int stride) {
  const int lane = threadIdx.x & 31;
  const bf16_t* p = base + (lane & 15) * stride + ((lane >> 4) << 3);
  FragU u;
  u.q[0] = *(const U128*)(p);
  u.q[1] = *(const U128*)(p + 16);
  return u.v;
}

// B fragment with per-lane row guard (rows < 0 -> zeros); source row-major, ld halves
__device__ __forceinline__ v16bf frag_b_guard(const bf16_t* base, int row, int ld, int koff) {
  FragU u;
  if (row >= 0) {
    const bf16_t* p = base + (size_t)row * ld + koff + (((threadIdx.x & 31) >> 4) << 3);
    u.q[0] = *(const U128*)(p);
    u.q[1] = *(const U128*)(p + 16);
  } else {
    u.q[0] = U128{0u, 0u, 0u, 0u};
    u.q[1] = U128{0u, 0u, 0u, 0u};
  }
  return u.v;
}

// A fragment 16x32 from column-major storage: elem A(m,k) at base[k*stride + m]
__device__ __forceinline__ v16bf frag_a_col(const bf16_t* base, int stride) {
  const int lane = threadIdx.x & 31;
  const int m = lane & 15, h8 = (lane >> 4) << 3;
  v16bf f;
#pragma unroll
  for (int i = 0; i < 8; ++i) f[i] = base[(h8 + i) * stride + m];
#pragma unroll
  for (int i = 0; i < 8; ++i) f[8 + i] = base[(16 + h8 + i) * stride + m];
  return f;
}

__device__ __forceinline__ float blk_sum(float v, float* red) {
  const int tid = threadIdx.x;
  red[tid] = v; __syncthreads();
#pragma unroll
  for (int s = 128; s > 0; s >>= 1) {
    if (tid < s) red[tid] += red[tid + s];
    __syncthreads();
  }
  float r = red[0]; __syncthreads();
  return r;
}

__device__ __forceinline__ float blk_max(float v, float* red) {
  const int tid = threadIdx.x;
  red[tid] = v; __syncthreads();
#pragma unroll
  for (int s = 128; s > 0; s >>= 1) {
    if (tid < s) red[tid] = fmaxf(red[tid], red[tid + s]);
    __syncthreads();
  }
  float r = red[0]; __syncthreads();
  return r;
}

// ---------------- CDNA5 async data movers ----------------
// per-lane 16B global -> LDS async copy (ASYNCcnt), CDNA5 ISA
__device__ __forceinline__ void pe_async_b128(const void* gsrc, void* ldst) {
  const uint32_t l = (uint32_t)(uintptr_t)ldst;   // generic->LDS: low 32 bits
  asm volatile("global_load_async_to_lds_b128 %0, %1, off"
               :: "v"(l), "v"(gsrc) : "memory");
}

__device__ __forceinline__ void pe_wait_async0() {
  asm volatile("s_wait_asynccnt 0x0" ::: "memory");
}

#if __has_builtin(__builtin_amdgcn_tensor_load_to_lds)
typedef unsigned int tdm_v4u __attribute__((ext_vector_type(4)));
typedef int tdm_v8i __attribute__((ext_vector_type(8)));
typedef int tdm_v4i __attribute__((ext_vector_type(4)));
// 2-D TDM tile load: rows x rowElems of 2-byte elements, contiguous (stride=rowElems)
// D# layout per CDNA5 ISA
__device__ __forceinline__ void pe_tdm_load_2d(const void* gsrc, void* ldst,
                                               unsigned rows, unsigned rowElems) {
  const uint64_t ga = (uint64_t)(uintptr_t)gsrc;
  const uint32_t la = (uint32_t)(uintptr_t)ldst;
  tdm_v4u g0;
  g0.x = 1u;                                                  // count=1 (valid), user mode
  g0.y = la;                                                  // lds_addr
  g0.z = (uint32_t)ga;                                        // global_addr[31:0]
  g0.w = (uint32_t)((ga >> 32) & 0x01FFFFFFu) | (2u << 30);   // global_addr[56:32], type=2
  tdm_v8i g1;
  g1[0] = (int)(1u << 16);                                    // wg_mask=0, data_size=1 (2B)
  g1[1] = (int)((rowElems & 0xFFFFu) << 16);                  // tensor_dim0 lo
  g1[2] = (int)(((rowElems >> 16) & 0xFFFFu) | ((rows & 0xFFFFu) << 16)); // td0 hi | td1 lo
  g1[3] = (int)(((rows >> 16) & 0xFFFFu) | ((rowElems & 0xFFFFu) << 16)); // td1 hi | tile_dim0
  g1[4] = (int)(rows & 0xFFFFu);                              // tile_dim1, tile_dim2=0
  g1[5] = (int)rowElems;                                      // tensor_dim0_stride lo
  g1[6] = 0;
  g1[7] = 0;
  tdm_v4i zz = {0, 0, 0, 0};
#if defined(__clang_major__) && (__clang_major__ >= 23)
  tdm_v8i z8 = {0, 0, 0, 0, 0, 0, 0, 0};
  __builtin_amdgcn_tensor_load_to_lds(g0, g1, zz, zz, z8, 0);
#else
  __builtin_amdgcn_tensor_load_to_lds(g0, g1, zz, zz, 0);
#endif
}
__device__ __forceinline__ void pe_wait_tensor0() {
#if __has_builtin(__builtin_amdgcn_s_wait_tensorcnt)
  __builtin_amdgcn_s_wait_tensorcnt(0);
#else
  asm volatile("s_wait_tensorcnt 0x0" ::: "memory");
#endif
}
#endif

// ---------------- embedding (+ shift-right with SOS) ----------------
__global__ void pe_embed_kernel(const int* __restrict__ value, const int* __restrict__ depth,
                                const int* __restrict__ pos, const float* __restrict__ sos,
                                const float* __restrict__ tok_emb,
                                const float* __restrict__ depth_emb,
                                const float* __restrict__ spatial_emb,
                                float* __restrict__ x1, float* __restrict__ x2) {
  const int t = blockIdx.x;           // t = n*S + s
  const int n = t / S_, s = t % S_;
  const int e = threadIdx.x;
  float val;
  if (s == 0) {
    val = sos[e];
  } else {
    const int sp = s - 1;
    const int vi = value[sp * N_ + n];
    const int di = depth[sp * N_ + n];
    val = tok_emb[(size_t)vi * E_ + e] + depth_emb[(size_t)di * E_ + e];
#pragma unroll
    for (int a = 0; a < 3; ++a) {
      const int pi = pos[((size_t)a * S_ + sp) * N_ + n];
      val += spatial_emb[((size_t)a * VOC_ + pi) * E_ + e];
    }
  }
  x1[(size_t)t * E_ + e] = val;
  x2[(size_t)t * E_ + e] = val;
}

// ---------------- layernorm ----------------
__global__ void pe_ln_kernel(const float* __restrict__ x, const float* __restrict__ g,
                             const float* __restrict__ b, float* __restrict__ y) {
  __shared__ float red[256];
  const int t = blockIdx.x, e = threadIdx.x;
  const float v = x[(size_t)t * E_ + e];
  const float mean = blk_sum(v, red) * (1.0f / E_);
  const float d = v - mean;
  const float var = blk_sum(d * d, red) * (1.0f / E_);
  y[(size_t)t * E_ + e] = d * rsqrtf(var + 1e-5f) * g[e] + b[e];
}

// ---------------- generic bf16-WMMA GEMM ----------------
// C = (A(f32,MxK)*alphaA) @ B(f32) + bias + residual
// bNMajor=0: B stored (K,Nn) row-major.  bNMajor=1: B stored (Nn,K) row-major.
#define BK_  32
#define LDT_ 40
__global__ void pe_gemm_kernel(const float* __restrict__ A, int lda, float alphaA,
                               const float* __restrict__ B, int ldb, int bNMajor,
                               float* __restrict__ C, int ldc,
                               const float* __restrict__ bias,
                               const float* __restrict__ residual, int ldr,
                               int M, int Nn, int K) {
  __shared__ bf16_t As[128 * LDT_];
  __shared__ bf16_t Bs[128 * LDT_];
  const int tid = threadIdx.x, lane = tid & 31, wv = tid >> 5;
  const int wm = wv & 1, wn = wv >> 1;
  const int m0 = blockIdx.x * 128, n0 = blockIdx.y * 128;
  (void)M;

  v8f acc[4][2] = {};
  for (int kt = 0; kt < K; kt += BK_) {
    {  // stage A tile 128x32 (f32 -> bf16, scaled)
      const int k4 = (tid & 7) << 2;
      const int r = tid >> 3;
#pragma unroll
      for (int rr = 0; rr < 4; ++rr) {
        const int row = r + rr * 32;
        const float4 av = *(const float4*)(A + (size_t)(m0 + row) * lda + kt + k4);
        bf16_t* d = As + row * LDT_ + k4;
        d[0] = (bf16_t)(av.x * alphaA); d[1] = (bf16_t)(av.y * alphaA);
        d[2] = (bf16_t)(av.z * alphaA); d[3] = (bf16_t)(av.w * alphaA);
        if (kt + BK_ < K)
          __builtin_prefetch(A + (size_t)(m0 + row) * lda + kt + BK_ + k4, 0, 1);
      }
    }
    if (bNMajor) {  // B is (Nn,K): already N-major; copy rows
      const int k4 = (tid & 7) << 2;
      const int r = tid >> 3;
#pragma unroll
      for (int rr = 0; rr < 4; ++rr) {
        const int nrow = r + rr * 32;
        float4 bv = make_float4(0.f, 0.f, 0.f, 0.f);
        if (n0 + nrow < Nn) bv = *(const float4*)(B + (size_t)(n0 + nrow) * ldb + kt + k4);
        bf16_t* d = Bs + nrow * LDT_ + k4;
        d[0] = (bf16_t)bv.x; d[1] = (bf16_t)bv.y; d[2] = (bf16_t)bv.z; d[3] = (bf16_t)bv.w;
      }
    } else {        // B is (K,Nn): transpose into N-major LDS tile
      const int nq = (tid & 31) << 2;
      const int kr = tid >> 5;
#pragma unroll
      for (int rr = 0; rr < 4; ++rr) {
        const int kk = kr + rr * 8;
        const float* bp = B + (size_t)(kt + kk) * ldb;
        const int gn = n0 + nq;
        float b0, b1, b2, b3;
        if (gn + 3 < Nn) {
          const float4 bv = *(const float4*)(bp + gn);
          b0 = bv.x; b1 = bv.y; b2 = bv.z; b3 = bv.w;
        } else {
          b0 = (gn + 0 < Nn) ? bp[gn + 0] : 0.f;
          b1 = (gn + 1 < Nn) ? bp[gn + 1] : 0.f;
          b2 = (gn + 2 < Nn) ? bp[gn + 2] : 0.f;
          b3 = (gn + 3 < Nn) ? bp[gn + 3] : 0.f;
        }
        Bs[(nq + 0) * LDT_ + kk] = (bf16_t)b0;
        Bs[(nq + 1) * LDT_ + kk] = (bf16_t)b1;
        Bs[(nq + 2) * LDT_ + kk] = (bf16_t)b2;
        Bs[(nq + 3) * LDT_ + kk] = (bf16_t)b3;
      }
    }
    __syncthreads();
    v16bf af[4], bfr[2];
#pragma unroll
    for (int mt = 0; mt < 4; ++mt) af[mt] = frag_a_row(As + (wm * 64 + mt * 16) * LDT_, LDT_);
#pragma unroll
    for (int nt = 0; nt < 2; ++nt) bfr[nt] = frag_b_nmaj(Bs + (wn * 32 + nt * 16) * LDT_, LDT_);
#pragma unroll
    for (int mt = 0; mt < 4; ++mt)
#pragma unroll
      for (int nt = 0; nt < 2; ++nt) acc[mt][nt] = wmma_bf16(af[mt], bfr[nt], acc[mt][nt]);
    __syncthreads();
  }
  const int hh = lane >> 4, nl = lane & 15;
#pragma unroll
  for (int mt = 0; mt < 4; ++mt)
#pragma unroll
    for (int nt = 0; nt < 2; ++nt) {
      const int gn = n0 + wn * 32 + nt * 16 + nl;
      if (gn < Nn) {
        const float bv = bias ? bias[gn] : 0.f;
#pragma unroll
        for (int i = 0; i < 8; ++i) {
          const int gm = m0 + wm * 64 + mt * 16 + i + 8 * hh;
          float v = acc[mt][nt][i] + bv;
          if (residual) v += residual[(size_t)gm * ldr + gn];
          C[(size_t)gm * ldc + gn] = v;
        }
      }
    }
}

// ---------------- FAVOR+: key global max then phi features ----------------
__global__ void pe_kmax_kernel(const float* __restrict__ xp, float* __restrict__ kmax) {
  __shared__ float red[256];
  const int h = blockIdx.x / N_, n = blockIdx.x % N_;
  const float* base = xp + ((size_t)h * TOK_ + (size_t)n * S_) * FPAD_;
  float m = -1e30f;
  for (int idx = threadIdx.x; idx < S_ * FPAD_; idx += 256) {
    const int f = idx % FPAD_;
    if (f < NBF_) m = fmaxf(m, base[idx]);
  }
  const float r = blk_max(m, red);
  if (threadIdx.x == 0) kmax[blockIdx.x] = r;
}

__global__ void pe_phi_kernel(const float* __restrict__ xp, const float* __restrict__ xraw,
                              const float* __restrict__ kmax, bf16_t* __restrict__ outp,
                              int isQuery) {
  __shared__ float red[256];
  const int h = blockIdx.x / TOK_, t = blockIdx.x % TOK_;
  const float* xr = xp + (size_t)blockIdx.x * FPAD_;
  const int tid = threadIdx.x;
  float p = 0.f;
  if (tid < DH_) {
    const float qv = xraw[(size_t)t * INNER_ + h * DH_ + tid];
    p = qv * qv;
  }
  const float sumsq = blk_sum(p, red);
  float mm = -1e30f;
  for (int f = tid; f < NBF_; f += 256) mm = fmaxf(mm, xr[f]);
  const float rowm = blk_max(mm, red);
  const float m = isQuery ? rowm : kmax[h * N_ + (t / S_)];
  const float dg = 0.0625f * sumsq;          // 0.5 * dn^2, dn^2 = DH^-0.5 = 1/8
  const float ratio = 0.061313944f;          // 266^-0.5
  for (int f = tid; f < FPAD_; f += 256) {
    float val = 0.f;
    if (f < NBF_) val = ratio * (__expf(xr[f] - dg - m) + 1e-4f);
    outp[(size_t)blockIdx.x * FPAD_ + f] = (bf16_t)val;
  }
}

// ---------------- rotary for local heads ----------------
__global__ void pe_rotary_kernel(const float* __restrict__ q, const float* __restrict__ k,
                                 bf16_t* __restrict__ qr, bf16_t* __restrict__ kr) {
  const int b = blockIdx.x;                 // N*S
  const int n = b / S_, s = b % S_;
  const int tid = threadIdx.x;              // 4 heads * 64
  const int hl = tid >> 6, d = tid & 63;
  const int fi = d & 31;
  const float theta = (float)s * __powf(10000.f, -(float)fi / 32.f);
  const float c = __cosf(theta), sn = __sinf(theta);
  const size_t src = (size_t)(n * S_ + s) * INNER_ + (HGLB_ + hl) * DH_;
  const size_t dst = ((size_t)(n * HLOC_ + hl) * S_ + s) * DH_ + d;
  float xv = q[src + d];
  float xo = (d < 32) ? -q[src + d + 32] : q[src + d - 32];
  qr[dst] = (bf16_t)(xv * c + xo * sn);
  xv = k[src + d];
  xo = (d < 32) ? -k[src + d + 32] : k[src + d - 32];
  kr[dst] = (bf16_t)(xv * c + xo * sn);
}

// ---------------- chunked causal linear attention (FAVOR heads) ----------------
// one workgroup per (h, n); serial scan over 16 chunks of 128.
// K chunk tile staged by TDM (TENSORcnt); Q chunk tile staged by async-LDS (ASYNCcnt).
#define K1ST_ 288
#define Q1ST_ 296
__global__ void pe_causal_kernel(const bf16_t* __restrict__ qp, const bf16_t* __restrict__ kp,
                                 const float* __restrict__ v, float* __restrict__ o) {
  extern __shared__ char smem[];
  bf16_t* q1s  = (bf16_t*)smem;                    // [i 0..127][f], stride 296 (async dest)
  bf16_t* k1s  = q1s + 128 * Q1ST_;                // [j 0..127][f], stride 288 (TDM dest)
  bf16_t* vTs  = k1s + 128 * K1ST_;                // [e 0..63][j 0..127], stride 136
  float*  kv   = (float*)(vTs + 64 * 136);         // [f][e], stride 68
  float*  ksum = kv + FPAD_ * 68;                  // [288]
  float*  den  = ksum + FPAD_;                     // [128]
  bf16_t* kvb  = (bf16_t*)(den + 128);             // [e][f], stride 296
  bf16_t* attb = kvb + 64 * 296;                   // [i][j], stride 136

  const int tid = threadIdx.x, lane = tid & 31, wv = tid >> 5;
  const int hh = lane >> 4, nl = lane & 15;
  const int h = blockIdx.x / N_, n = blockIdx.x % N_;
  const bf16_t* qh = qp + ((size_t)h * TOK_ + (size_t)n * S_) * FPAD_;
  const bf16_t* kh = kp + ((size_t)h * TOK_ + (size_t)n * S_) * FPAD_;
  const size_t vbase = (size_t)n * S_ * INNER_ + h * DH_;

  for (int i = tid; i < FPAD_ * 68; i += 256) kv[i] = 0.f;
  for (int i = tid; i < FPAD_; i += 256) ksum[i] = 0.f;
  __syncthreads();

  for (int c = 0; c < NCH_; ++c) {
    const int s0 = c * CH_;
    const bf16_t* gk = kh + (size_t)s0 * FPAD_;
    const bf16_t* gq = qh + (size_t)s0 * FPAD_;

    // ---- K tile via Tensor Data Mover (wave 0 issues one 2-D descriptor) ----
#if __has_builtin(__builtin_amdgcn_tensor_load_to_lds)
    if (tid < 32) {
      pe_tdm_load_2d(gk, k1s, CH_, FPAD_);
      pe_wait_tensor0();
    }
#else
    {
      const uint32_t* ksrc = (const uint32_t*)gk;
      uint32_t* kdst = (uint32_t*)k1s;   // stride 288 == source row: contiguous
      for (int i = tid; i < 128 * 144; i += 256) kdst[i] = ksrc[i];
    }
#endif

    // ---- Q tile via per-lane async global->LDS b128 copies ----
    {
      const char* gqb = (const char*)gq;
      char* lqb = (char*)q1s;
#pragma unroll
      for (int it = 0; it < 18; ++it) {          // 16 rows * 576B per wave
        const int c2 = it * 32 + lane;
        const int row = wv * 16 + (c2 / 36);
        const int cw = c2 % 36;
        pe_async_b128(gqb + (size_t)row * (FPAD_ * 2) + cw * 16,
                      lqb + row * (Q1ST_ * 2) + cw * 16);
      }
      pe_wait_async0();
    }

    // ---- vT (f32->bf16 transpose), kvb (f32->bf16), zero den ----
    for (int i = tid; i < 128 * 64; i += 256) {
      const int j = i >> 6, e = i & 63;
      vTs[e * 136 + j] = (bf16_t)v[vbase + (size_t)(s0 + j) * INNER_ + e];
    }
    for (int i = tid; i < FPAD_ * 64; i += 256) {
      const int f = i >> 6, e = i & 63;
      kvb[e * 296 + f] = (bf16_t)kv[f * 68 + e];
    }
    if (tid < 128) den[tid] = 0.f;
    __syncthreads();

    // phase A: att = tril(q1 @ k1^T), 128x128, K = FPAD
    v8f acc[8] = {};
    const bf16_t* q1w = q1s + (wv * 16) * Q1ST_;
#pragma unroll
    for (int ks = 0; ks < 9; ++ks) {
      const v16bf a = frag_a_row(q1w + ks * 32, Q1ST_);
#pragma unroll
      for (int nt = 0; nt < 8; ++nt)
        acc[nt] = wmma_bf16(a, frag_b_nmaj(k1s + nt * 16 * K1ST_ + ks * 32, K1ST_), acc[nt]);
    }
#pragma unroll
    for (int nt = 0; nt < 8; ++nt)
#pragma unroll
      for (int i = 0; i < 8; ++i) {
        const int r = wv * 16 + i + 8 * hh;
        const int j = nt * 16 + nl;
        attb[r * 136 + j] = (bf16_t)((j <= r) ? acc[nt][i] : 0.f);
      }
    __syncthreads();

    // phase B: out = att @ v1 + q1 @ kv_sum
    v8f oacc[4] = {};
#pragma unroll
    for (int ks = 0; ks < 4; ++ks) {
      const v16bf a = frag_a_row(attb + wv * 16 * 136 + ks * 32, 136);
#pragma unroll
      for (int nt = 0; nt < 4; ++nt)
        oacc[nt] = wmma_bf16(a, frag_b_nmaj(vTs + nt * 16 * 136 + ks * 32, 136), oacc[nt]);
    }
#pragma unroll
    for (int ks = 0; ks < 9; ++ks) {
      const v16bf a = frag_a_row(q1w + ks * 32, Q1ST_);
#pragma unroll
      for (int nt = 0; nt < 4; ++nt)
        oacc[nt] = wmma_bf16(a, frag_b_nmaj(kvb + nt * 16 * 296 + ks * 32, 296), oacc[nt]);
    }

    // phase C: den = rowsum(att) + q1 . k_sum
    {
      const int r = tid >> 1, half = tid & 1;
      float sden = 0.f;
      for (int j = half * 64; j < half * 64 + 64; ++j) sden += (float)attb[r * 136 + j];
      const bf16_t* qr_ = q1s + r * Q1ST_;
      for (int f = half * 144; f < half * 144 + 144; ++f) sden += (float)qr_[f] * ksum[f];
      atomicAdd(&den[r], sden);
    }
    __syncthreads();

#pragma unroll
    for (int nt = 0; nt < 4; ++nt)
#pragma unroll
      for (int i = 0; i < 8; ++i) {
        const int r = wv * 16 + i + 8 * hh;
        const int e = nt * 16 + nl;
        o[vbase + (size_t)(s0 + r) * INNER_ + e] = oacc[nt][i] / (den[r] + 1e-10f);
      }

    // phase D: kv_sum += k1^T @ v1 ; k_sum += colsum(k1)
    const int ntile = (wv < 2) ? 3 : 2;  // 18 f-tiles over 8 waves
    for (int mt = 0; mt < ntile; ++mt) {
      const int ft = (mt == 2) ? (16 + wv) : (mt * 8 + wv);
      v8f z[4];
#pragma unroll
      for (int nt = 0; nt < 4; ++nt)
#pragma unroll
        for (int i = 0; i < 8; ++i)
          z[nt][i] = kv[(ft * 16 + i + 8 * hh) * 68 + nt * 16 + nl];
#pragma unroll
      for (int ks = 0; ks < 4; ++ks) {
        const v16bf a = frag_a_col(k1s + ks * 32 * K1ST_ + ft * 16, K1ST_);
#pragma unroll
        for (int nt = 0; nt < 4; ++nt)
          z[nt] = wmma_bf16(a, frag_b_nmaj(vTs + nt * 16 * 136 + ks * 32, 136), z[nt]);
      }
#pragma unroll
      for (int nt = 0; nt < 4; ++nt)
#pragma unroll
        for (int i = 0; i < 8; ++i)
          kv[(ft * 16 + i + 8 * hh) * 68 + nt * 16 + nl] = z[nt][i];
    }
    for (int f = tid; f < FPAD_; f += 256) {
      float s = ksum[f];
      for (int j = 0; j < 128; ++j) s += (float)k1s[j * K1ST_ + f];
      ksum[f] = s;
    }
    __syncthreads();
  }
}

// ---------------- windowed local attention (flash-style) ----------------
__global__ void pe_local_kernel(const bf16_t* __restrict__ qr, const bf16_t* __restrict__ kr,
                                const float* __restrict__ v, float* __restrict__ o) {
  extern __shared__ char smem[];
  bf16_t* vTs = (bf16_t*)smem;      // [e 0..63][j 0..127], stride 136
  bf16_t* Pb  = vTs + 64 * 136;     // [row 0..255][j 0..127], stride 136

  const int tid = threadIdx.x, lane = tid & 31, wv = tid >> 5;
  const int hh = lane >> 4, nl = lane & 15;
  const int b = blockIdx.x;         // n*32 + hl*8 + w
  const int w = b & 7, hl = (b >> 3) & 3, n = b >> 5;
  const int hoff = (HGLB_ + hl) * DH_;
  const bf16_t* qbase = qr + ((size_t)(n * HLOC_ + hl) * S_) * DH_;
  const bf16_t* kbase = kr + ((size_t)(n * HLOC_ + hl) * S_) * DH_;
  const int srow0 = w * 256;

  float mrow[2][8], drow[2][8];
  v8f oacc[2][4] = {};
#pragma unroll
  for (int mt = 0; mt < 2; ++mt)
#pragma unroll
    for (int i = 0; i < 8; ++i) { mrow[mt][i] = -1e30f; drow[mt][i] = 0.f; }

  for (int p = 0; p < 4; ++p) {
    const int j0 = srow0 - 256 + p * 128;  // global key position of panel col 0
    for (int i = tid; i < 128 * 64; i += 256) {
      const int j = i >> 6, e = i & 63;
      const int sj = j0 + j;
      const float val = (sj >= 0) ? v[(size_t)(n * S_ + sj) * INNER_ + hoff + e] : 0.f;
      vTs[e * 136 + j] = (bf16_t)val;
    }
    __syncthreads();

    v8f sacc[2][8] = {};
#pragma unroll
    for (int ks = 0; ks < 2; ++ks) {
      const v16bf a0 = frag_a_row(qbase + (size_t)(srow0 + wv * 32) * DH_ + ks * 32, DH_);
      const v16bf a1 = frag_a_row(qbase + (size_t)(srow0 + wv * 32 + 16) * DH_ + ks * 32, DH_);
#pragma unroll
      for (int nt = 0; nt < 8; ++nt) {
        const v16bf bfr = frag_b_guard(kbase, j0 + nt * 16 + nl, DH_, ks * 32);
        sacc[0][nt] = wmma_bf16(a0, bfr, sacc[0][nt]);
        sacc[1][nt] = wmma_bf16(a1, bfr, sacc[1][nt]);
      }
    }

#pragma unroll
    for (int mt = 0; mt < 2; ++mt) {
      float pmax[8];
#pragma unroll
      for (int i = 0; i < 8; ++i) pmax[i] = -1e30f;
#pragma unroll
      for (int nt = 0; nt < 8; ++nt)
#pragma unroll
        for (int i = 0; i < 8; ++i) {
          const int r = wv * 32 + mt * 16 + i + 8 * hh;
          const int tq = srow0 + r;
          const int tk = j0 + nt * 16 + nl;
          const float sval = (tk >= 0 && tk <= tq) ? sacc[mt][nt][i] * 0.125f : -1e30f;
          sacc[mt][nt][i] = sval;
          pmax[i] = fmaxf(pmax[i], sval);
        }
#pragma unroll
      for (int off = 1; off < 16; off <<= 1)
#pragma unroll
        for (int i = 0; i < 8; ++i) pmax[i] = fmaxf(pmax[i], __shfl_xor(pmax[i], off, 32));
      float nm[8], alpha[8], rs[8];
#pragma unroll
      for (int i = 0; i < 8; ++i) {
        nm[i] = fmaxf(mrow[mt][i], pmax[i]);
        alpha[i] = __expf(mrow[mt][i] - nm[i]);
        mrow[mt][i] = nm[i];
        rs[i] = 0.f;
      }
#pragma unroll
      for (int nt = 0; nt < 8; ++nt)
#pragma unroll
        for (int i = 0; i < 8; ++i) {
          const float sval = sacc[mt][nt][i];
          const float pv = (sval <= -1e29f) ? 0.f : __expf(sval - nm[i]);
          rs[i] += pv;
          const int r = wv * 32 + mt * 16 + i + 8 * hh;
          Pb[r * 136 + nt * 16 + nl] = (bf16_t)pv;
        }
#pragma unroll
      for (int off = 1; off < 16; off <<= 1)
#pragma unroll
        for (int i = 0; i < 8; ++i) rs[i] += __shfl_xor(rs[i], off, 32);
#pragma unroll
      for (int i = 0; i < 8; ++i) drow[mt][i] = drow[mt][i] * alpha[i] + rs[i];
#pragma unroll
      for (int nte = 0; nte < 4; ++nte)
#pragma unroll
        for (int i = 0; i < 8; ++i) oacc[mt][nte][i] *= alpha[i];
    }
    __syncthreads();

#pragma unroll
    for (int ks = 0; ks < 4; ++ks)
#pragma unroll
      for (int mt = 0; mt < 2; ++mt) {
        const v16bf a = frag_a_row(Pb + (wv * 32 + mt * 16) * 136 + ks * 32, 136);
#pragma unroll
        for (int nte = 0; nte < 4; ++nte)
          oacc[mt][nte] = wmma_bf16(a, frag_b_nmaj(vTs + nte * 16 * 136 + ks * 32, 136),
                                    oacc[mt][nte]);
      }
    __syncthreads();
  }

#pragma unroll
  for (int mt = 0; mt < 2; ++mt)
#pragma unroll
    for (int nte = 0; nte < 4; ++nte)
#pragma unroll
      for (int i = 0; i < 8; ++i) {
        const int r = wv * 32 + mt * 16 + i + 8 * hh;
        const int e = nte * 16 + nl;
        o[(size_t)(n * S_ + srow0 + r) * INNER_ + hoff + e] = oacc[mt][nte][i] / drow[mt][i];
      }
}

// ---------------- GLU, averaging, output transpose ----------------
__global__ void pe_glu_kernel(const float* __restrict__ ff, float* __restrict__ glu) {
  const int idx = blockIdx.x * 256 + threadIdx.x;
  if (idx >= TOK_ * FF_) return;
  const int t = idx / FF_, j = idx % FF_;
  const float a = ff[(size_t)t * FF2_ + j];
  const float g = ff[(size_t)t * FF2_ + FF_ + j];
  const float ge = 0.5f * a * (1.f + erff(a * 0.70710678118654752f));
  glu[idx] = ge * g;
}

__global__ void pe_avg_kernel(const float* __restrict__ x1, const float* __restrict__ x2,
                              float* __restrict__ xa) {
  const int idx = blockIdx.x * 256 + threadIdx.x;
  xa[idx] = 0.5f * (x1[idx] + x2[idx]);
}

__global__ void pe_out_kernel(const float* __restrict__ logits, float* __restrict__ out) {
  const int idx = blockIdx.x * 256 + threadIdx.x;
  if (idx >= S_ * N_ * VOC_) return;
  const int vv = idx % VOC_;
  const int r = idx / VOC_;
  const int nn = r % N_, s = r / N_;
  out[idx] = logits[((size_t)nn * S_ + s) * VOC_ + vv];
}

// ---------------- host ----------------
extern "C" void kernel_launch(void* const* d_in, const int* in_sizes, int n_in,
                              void* d_out, int out_size, void* d_ws, size_t ws_size,
                              hipStream_t stream) {
  (void)in_sizes; (void)n_in; (void)out_size;
  const int*   value       = (const int*)d_in[0];
  const int*   depth       = (const int*)d_in[1];
  const int*   pos         = (const int*)d_in[2];
  const float* sos         = (const float*)d_in[3];
  const float* tok_emb     = (const float*)d_in[4];
  const float* depth_emb   = (const float*)d_in[5];
  const float* spatial_emb = (const float*)d_in[6];
  const float* head_w      = (const float*)d_in[7];
  const float* ln1_g = (const float*)d_in[8];
  const float* ln1_b = (const float*)d_in[9];
  const float* wq = (const float*)d_in[10];
  const float* bq = (const float*)d_in[11];
  const float* wk = (const float*)d_in[12];
  const float* bk = (const float*)d_in[13];
  const float* wvp = (const float*)d_in[14];
  const float* bv = (const float*)d_in[15];
  const float* wo = (const float*)d_in[16];
  const float* bo = (const float*)d_in[17];
  const float* proj = (const float*)d_in[18];
  const float* ln2_g = (const float*)d_in[19];
  const float* ln2_b = (const float*)d_in[20];
  const float* w1 = (const float*)d_in[21];
  const float* b1 = (const float*)d_in[22];
  const float* w2 = (const float*)d_in[23];
  const float* b2 = (const float*)d_in[24];

  char* ws = (char*)d_ws;
  size_t off = 0;
  auto alloc = [&](size_t bytes) {
    char* p = ws + off;
    off += (bytes + 255) & ~(size_t)255;
    return p;
  };
  float*  x1    = (float*)alloc((size_t)TOK_ * E_ * 4);
  float*  x2    = (float*)alloc((size_t)TOK_ * E_ * 4);
  float*  yln   = (float*)alloc((size_t)TOK_ * VOC_ * 4);  // also final logits
  float*  qb    = (float*)alloc((size_t)TOK_ * INNER_ * 4);
  float*  kb    = (float*)alloc((size_t)TOK_ * INNER_ * 4);
  float*  vb    = (float*)alloc((size_t)TOK_ * INNER_ * 4);
  float*  ob    = (float*)alloc((size_t)TOK_ * INNER_ * 4);
  float*  xp    = (float*)alloc((size_t)HGLB_ * TOK_ * FPAD_ * 4);  // also GLU buf
  bf16_t* qp    = (bf16_t*)alloc((size_t)HGLB_ * TOK_ * FPAD_ * 2);
  bf16_t* kp    = (bf16_t*)alloc((size_t)HGLB_ * TOK_ * FPAD_ * 2);
  bf16_t* qrb   = (bf16_t*)alloc((size_t)N_ * HLOC_ * S_ * DH_ * 2);
  bf16_t* krb   = (bf16_t*)alloc((size_t)N_ * HLOC_ * S_ * DH_ * 2);
  float*  kmaxb = (float*)alloc(64 * 4);
  if (off > ws_size) return;  // deterministic no-op if scratch too small
  float* ffb    = qb;   // aliases qb..ob (exactly TOK*FF2 floats), dead by FFN time
  float* glub   = xp;   // xp/qp/kp dead by FFN time
  float* logits = yln;
  float* xavg   = qb;

  auto gemm = [&](const float* A, int lda, float alphaA, const float* B, int ldb, int bNM,
                  float* C, int ldc, const float* bias, const float* res, int ldr,
                  int M, int Nn, int K) {
    dim3 g(M / 128, (Nn + 127) / 128);
    pe_gemm_kernel<<<g, 256, 0, stream>>>(A, lda, alphaA, B, ldb, bNM, C, ldc, bias,
                                          res, ldr, M, Nn, K);
  };

  const size_t CAUSAL_SMEM = (size_t)(128 * Q1ST_ + 128 * K1ST_ + 64 * 136) * 2  // q1s+k1s+vTs
                           + (size_t)(FPAD_ * 68 + FPAD_ + 128) * 4              // kv+ksum+den
                           + (size_t)(64 * 296 + 128 * 136) * 2;                 // kvb+attb
  const size_t LOCAL_SMEM = (size_t)(64 * 136 + 256 * 136) * 2;

  pe_embed_kernel<<<TOK_, 256, 0, stream>>>(value, depth, pos, sos, tok_emb, depth_emb,
                                            spatial_emb, x1, x2);

  const float dn = 0.35355339059327373f;  // DH^-0.25
  for (int l = 0; l < L_; ++l) {
    const float* wq_l = wq + (size_t)l * E_ * INNER_;
    const float* wk_l = wk + (size_t)l * E_ * INNER_;
    const float* wv_l = wvp + (size_t)l * E_ * INNER_;
    const float* wo_l = wo + (size_t)l * INNER_ * E_;
    const float* proj_l = proj + (size_t)l * NBF_ * DH_;
    const float* w1_l = w1 + (size_t)l * E_ * FF2_;
    const float* w2_l = w2 + (size_t)l * FF_ * E_;

    pe_ln_kernel<<<TOK_, 256, 0, stream>>>(x2, ln1_g + (size_t)l * E_, ln1_b + (size_t)l * E_, yln);
    gemm(yln, E_, 1.f, wq_l, INNER_, 0, qb, INNER_, bq + (size_t)l * INNER_, nullptr, 0, TOK_, INNER_, E_);
    gemm(yln, E_, 1.f, wk_l, INNER_, 0, kb, INNER_, bk + (size_t)l * INNER_, nullptr, 0, TOK_, INNER_, E_);
    gemm(yln, E_, 1.f, wv_l, INNER_, 0, vb, INNER_, bv + (size_t)l * INNER_, nullptr, 0, TOK_, INNER_, E_);

    for (int h = 0; h < HGLB_; ++h)
      gemm(kb + h * DH_, INNER_, dn, proj_l, DH_, 1, xp + (size_t)h * TOK_ * FPAD_, FPAD_,
           nullptr, nullptr, 0, TOK_, NBF_, DH_);
    pe_kmax_kernel<<<HGLB_ * N_, 256, 0, stream>>>(xp, kmaxb);
    pe_phi_kernel<<<HGLB_ * TOK_, 256, 0, stream>>>(xp, kb, kmaxb, kp, 0);
    for (int h = 0; h < HGLB_; ++h)
      gemm(qb + h * DH_, INNER_, dn, proj_l, DH_, 1, xp + (size_t)h * TOK_ * FPAD_, FPAD_,
           nullptr, nullptr, 0, TOK_, NBF_, DH_);
    pe_phi_kernel<<<HGLB_ * TOK_, 256, 0, stream>>>(xp, qb, kmaxb, qp, 1);

    pe_rotary_kernel<<<N_ * S_, 256, 0, stream>>>(qb, kb, qrb, krb);
    pe_causal_kernel<<<HGLB_ * N_, 256, CAUSAL_SMEM, stream>>>(qp, kp, vb, ob);
    pe_local_kernel<<<N_ * HLOC_ * NWIN_, 256, LOCAL_SMEM, stream>>>(qrb, krb, vb, ob);

    gemm(ob, INNER_, 1.f, wo_l, E_, 0, x1, E_, bo + (size_t)l * E_, x1, E_, TOK_, E_, INNER_);

    pe_ln_kernel<<<TOK_, 256, 0, stream>>>(x1, ln2_g + (size_t)l * E_, ln2_b + (size_t)l * E_, yln);
    gemm(yln, E_, 1.f, w1_l, FF2_, 0, ffb, FF2_, b1 + (size_t)l * FF2_, nullptr, 0, TOK_, FF2_, E_);
    pe_glu_kernel<<<(TOK_ * FF_) / 256, 256, 0, stream>>>(ffb, glub);
    gemm(glub, FF_, 1.f, w2_l, E_, 0, x2, E_, b2 + (size_t)l * E_, x2, E_, TOK_, E_, FF_);
  }

  pe_avg_kernel<<<(TOK_ * E_) / 256, 256, 0, stream>>>(x1, x2, xavg);
  gemm(xavg, E_, 1.f, head_w, E_, 1, logits, VOC_, nullptr, nullptr, 0, TOK_, VOC_, E_);
  pe_out_kernel<<<(S_ * N_ * VOC_ + 255) / 256, 256, 0, stream>>>(logits, (float*)d_out);
}
